// FourierBlock_40278203302084
// MI455X (gfx1250) — matrix operand order repaired and checked by reading, not compiled, not run on hardware
//
#include <hip/hip_runtime.h>
#include <hip/hip_bf16.h>
#include <math.h>

typedef _Float16 half8  __attribute__((ext_vector_type(8)));
typedef _Float16 half16 __attribute__((ext_vector_type(16)));
typedef float    v8f    __attribute__((ext_vector_type(8)));

#define Bn 16
#define Ln 4096
#define Hn 8
#define En 64
#define On 64
#define Mn 64
#define ROWS (Bn*Hn*En)      // 8192
#define CCOL 128             // [cos | -sin] packed columns

// ---------------------------------------------------------------------------
// K0: build DFT basis matrices (f16) in workspace.
//   Ft[c][t]  (128 x 4096, K-contiguous per column) : c<64 -> cos(2pi m t/L),
//             c>=64 -> -sin(2pi m t/L).   (forward DFT, stage-1 B matrix)
//   Gt[t][c]  (4096 x 128, K-contiguous per column) : inverse basis with the
//             irfft scaling folded in: (m==0?1:2)/L * {cos | -sin}.
// ---------------------------------------------------------------------------
__global__ void k_basis(_Float16* __restrict__ Ft, _Float16* __restrict__ Gt) {
    int idx = blockIdx.x * 256 + threadIdx.x;      // 4096*128 total
    int t = idx >> 7;
    int c = idx & 127;
    int m = c & 63;
    int p = (m * t) & (Ln - 1);                    // (m*t) mod L
    float theta = (float)p * (6.28318530717958647692f / (float)Ln);
    float s, co;
    __sincosf(theta, &s, &co);
    float fwd = (c < 64) ? co : -s;
    float scale = ((m == 0) ? 1.0f : 2.0f) * (1.0f / (float)Ln);
    float inv = (c < 64) ? scale * co : -scale * s;
    Ft[(size_t)c * Ln + t] = (_Float16)fwd;
    Gt[(size_t)t * CCOL + c] = (_Float16)inv;
}

// ---------------------------------------------------------------------------
// K1: pack/transpose q (B,L,H,E f32) -> Apack[(b,h,e)][t] f16 (8192 x 4096),
// via LDS 64x64 tile so global reads (along e) and writes (along t) coalesce.
// ---------------------------------------------------------------------------
#define LDS_STRIDE 66
__global__ void k_pack(const float* __restrict__ q, _Float16* __restrict__ Apack) {
    __shared__ _Float16 tile[64 * LDS_STRIDE];
    int tid = threadIdx.x;
    int bi  = blockIdx.x;            // B*H*(L/64) = 8192 blocks
    int t0  = (bi & 63) * 64;
    int bh  = bi >> 6;               // b*8 + h
    int h   = bh & 7;
    int b   = bh >> 3;

    // load 64(t) x 64(e) f32 tile, convert to f16 into LDS[t][e]
    for (int r = 0; r < 4; ++r) {
        int tt = r * 16 + (tid >> 4);
        int e4 = (tid & 15) * 4;
        const float4 v = *(const float4*)(q + (((size_t)b * Ln + (t0 + tt)) * Hn + h) * En + e4);
        tile[tt * LDS_STRIDE + e4 + 0] = (_Float16)v.x;
        tile[tt * LDS_STRIDE + e4 + 1] = (_Float16)v.y;
        tile[tt * LDS_STRIDE + e4 + 2] = (_Float16)v.z;
        tile[tt * LDS_STRIDE + e4 + 3] = (_Float16)v.w;
    }
    __syncthreads();

    // write transposed: row (b,h,e), contiguous along t (16 halves/thread)
    int e  = tid >> 2;
    int tq = (tid & 3) * 16;
    size_t row = (size_t)bh * En + e;
    _Float16* dst = Apack + row * Ln + t0 + tq;
    half8 o0, o1;
    for (int j = 0; j < 8; ++j) {
        o0[j] = tile[(tq + j) * LDS_STRIDE + e];
        o1[j] = tile[(tq + 8 + j) * LDS_STRIDE + e];
    }
    *(half8*)(dst)     = o0;
    *(half8*)(dst + 8) = o1;
}

// ---------------------------------------------------------------------------
// WMMA fragment helpers (wave32, 16x16x32 f16):
//  A: lane<16 -> M=lane, kb=0 ; lane>=16 -> M=lane-16, kb=16
//     half16 elem j : K = kb + j
//     => two adjacent 16B chunks per lane: [kb..kb+7], [kb+8..kb+15]
//  B: lane<16 -> N=lane, K=0..15 ; lane>=16 -> N=lane-16, K=16..31
//     => one contiguous 32B chunk per lane.
//  C/D: n=lane&15, M = (lane<16?0:8)+r for VGPR r.
// ---------------------------------------------------------------------------
__device__ __forceinline__ half16 load_a_frag(const _Float16* rowptr, int kb) {
    half8 a0 = *(const half8*)(rowptr + kb);
    half8 a1 = *(const half8*)(rowptr + kb + 8);
    return __builtin_shufflevector(a0, a1, 0,1,2,3,4,5,6,7,8,9,10,11,12,13,14,15);
}

// ---------------------------------------------------------------------------
// K2: stage-1 GEMM:  X (8192 x 128 f32) = Apack (8192x4096) * Ft (4096x128)
// block = 8 waves; each block owns one 16-row tile; wave w owns 16-col tile w.
// ---------------------------------------------------------------------------
__global__ void k_gemm_fwd(const _Float16* __restrict__ Apack,
                           const _Float16* __restrict__ Ft,
                           float* __restrict__ X) {
    int tid  = threadIdx.x;
    int lane = tid & 31;
    int wid  = tid >> 5;             // 0..7 -> column tile
    int rowtile = blockIdx.x;        // 0..511

    int n  = lane & 15;
    int kb = (lane < 16) ? 0 : 16;   // A chunk offset
    int kB = (lane < 16) ? 0 : 16;   // B chunk offset

    const _Float16* Aptr = Apack + ((size_t)rowtile * 16 + n) * Ln;
    const _Float16* Bptr = Ft + ((size_t)(wid * 16 + n)) * Ln;

    v8f acc = {};
    for (int k0 = 0; k0 < Ln; k0 += 32) {
        __builtin_prefetch(Aptr + k0 + 256, 0, 0);   // global_prefetch_b8
        half16 a = load_a_frag(Aptr + k0, kb);
        half16 b = *(const half16*)(Bptr + k0 + kB);
        acc = __builtin_amdgcn_wmma_f32_16x16x32_f16(false, a, false, b,
                                                     (short)0, acc, false, false);
    }

    int mbase = (lane < 16) ? 0 : 8;
    float* Crow = X + ((size_t)rowtile * 16 + mbase) * CCOL + wid * 16 + n;
    for (int r = 0; r < 8; ++r)
        Crow[(size_t)r * CCOL] = acc[r];
}

// ---------------------------------------------------------------------------
// K3: mode mixing (complex, f32 VALU — only 0.27 GFLOP):
//   Yr[b,h,o,m] = sum_e Xr*Wr - Xi*Wi ;  Yi = sum_e Xr*Wi + Xi*Wr
// X rows: (b,h,e) x [m | 64+m].  Output packed f16 for stage-3 A fragments.
// ---------------------------------------------------------------------------
__global__ void k_mix(const float* __restrict__ X,
                      const float* __restrict__ Wr,
                      const float* __restrict__ Wi,
                      _Float16* __restrict__ Y) {
    int idx = blockIdx.x * 256 + threadIdx.x;   // B*H*O*M = 524288
    int m  = idx & 63;
    int o  = (idx >> 6) & 63;
    int bh = idx >> 12;                         // b*8 + h
    int h  = bh & 7;

    const float* Xb = X + (size_t)bh * En * CCOL;
    const float* wr = Wr + (size_t)h * En * On * Mn + o * Mn + m;
    const float* wi = Wi + (size_t)h * En * On * Mn + o * Mn + m;

    float yr = 0.0f, yi = 0.0f;
    for (int e = 0; e < En; ++e) {
        float xr = Xb[e * CCOL + m];
        float xi = Xb[e * CCOL + 64 + m];
        float a  = wr[(size_t)e * On * Mn];
        float bb = wi[(size_t)e * On * Mn];
        yr = fmaf(xr, a, fmaf(-xi, bb, yr));
        yi = fmaf(xr, bb, fmaf(xi, a, yi));
    }
    _Float16* out = Y + ((size_t)bh * On + o) * CCOL;
    out[m]      = (_Float16)yr;
    out[64 + m] = (_Float16)yi;
}

// ---------------------------------------------------------------------------
// K4: stage-3 GEMM: out (8192 x 4096 f32) = Y (8192x128) * Ginv (128x4096)
// Ginv stored column-K-contiguous as Gt[t][c]. grid = (32 colgroups, 512 rows)
// ---------------------------------------------------------------------------
__global__ void k_gemm_inv(const _Float16* __restrict__ Y,
                           const _Float16* __restrict__ Gt,
                           float* __restrict__ out) {
    int tid  = threadIdx.x;
    int lane = tid & 31;
    int wid  = tid >> 5;
    int rowtile = blockIdx.y;                    // 0..511
    int coltile = blockIdx.x * 8 + wid;          // 0..255

    int n  = lane & 15;
    int kb = (lane < 16) ? 0 : 16;
    int kB = (lane < 16) ? 0 : 16;

    const _Float16* Aptr = Y  + ((size_t)rowtile * 16 + n) * CCOL;
    const _Float16* Bptr = Gt + ((size_t)(coltile * 16 + n)) * CCOL;

    v8f acc = {};
    for (int k0 = 0; k0 < CCOL; k0 += 32) {
        half16 a = load_a_frag(Aptr + k0, kb);
        half16 b = *(const half16*)(Bptr + k0 + kB);
        acc = __builtin_amdgcn_wmma_f32_16x16x32_f16(false, a, false, b,
                                                     (short)0, acc, false, false);
    }

    int mbase = (lane < 16) ? 0 : 8;
    float* Crow = out + ((size_t)rowtile * 16 + mbase) * Ln + coltile * 16 + n;
    for (int r = 0; r < 8; ++r)
        Crow[(size_t)r * Ln] = acc[r];
}

// ---------------------------------------------------------------------------
extern "C" void kernel_launch(void* const* d_in, const int* in_sizes, int n_in,
                              void* d_out, int out_size, void* d_ws, size_t ws_size,
                              hipStream_t stream) {
    const float* q  = (const float*)d_in[0];
    const float* Wr = (const float*)d_in[4];
    const float* Wi = (const float*)d_in[5];
    float* out = (float*)d_out;

    // workspace carve-up (72 MB total)
    _Float16* Ft    = (_Float16*)d_ws;                       // 128*4096  (1 MB)
    _Float16* Gt    = Ft + (size_t)CCOL * Ln;                // 4096*128  (1 MB)
    _Float16* Apack = Gt + (size_t)Ln * CCOL;                // 8192*4096 (64 MB)
    float*    X     = (float*)(Apack + (size_t)ROWS * Ln);   // 8192*128  (4 MB)
    _Float16* Y     = (_Float16*)(X + (size_t)ROWS * CCOL);  // 8192*128  (2 MB)

    k_basis   <<<dim3((Ln * CCOL) / 256), dim3(256), 0, stream>>>(Ft, Gt);
    k_pack    <<<dim3(Bn * Hn * (Ln / 64)), dim3(256), 0, stream>>>(q, Apack);
    k_gemm_fwd<<<dim3(ROWS / 16), dim3(256), 0, stream>>>(Apack, Ft, X);
    k_mix     <<<dim3((Bn * Hn * On * Mn) / 256), dim3(256), 0, stream>>>(X, Wr, Wi, Y);
    k_gemm_inv<<<dim3(Ln / 128, ROWS / 16), dim3(256), 0, stream>>>(Y, Gt, out);
}
